// RNNModel_43568148251010
// MI455X (gfx1250) — matrix operand order repaired and checked by reading, not compiled, not run on hardware
//
#include <hip/hip_runtime.h>
#include <hip/hip_bf16.h>
#include <stdint.h>

// ---------------------------------------------------------------------------
// LSTM (B=512, I=128, T=1024, H=89, O=64) for gfx1250.
//   K0: prep   -> bf16 W_ih [368x128], bf16 W_hh [368x96] (zero padded),
//                 bias = b_ih+b_hh [368] f32.
//   K1: GEMM   -> xg[b,t,g] = x_bti . W_ih^T + bias, bf16 [B,T,368].
//                 v_wmma_f32_16x16x32_bf16; result staged in LDS and bulk
//                 stored with TDM tensor_store_from_lds (contiguous 11.5 KB).
//   K2: scan   -> per-WG 16 batch rows, T sequential steps:
//                 gates = xg + h.W_hh^T (WMMA), elementwise LSTM cell,
//                 xg tiles streamed HBM->LDS by TDM tensor_load_to_lds,
//                 double-buffered with s_wait_tensorcnt.  Fused W_out proj.
// ---------------------------------------------------------------------------

typedef __bf16 bf16_t;
typedef __attribute__((ext_vector_type(16))) __bf16   v16bf;
typedef __attribute__((ext_vector_type(8)))  __bf16   v8bf;
typedef __attribute__((ext_vector_type(8)))  float    v8f;
typedef __attribute__((ext_vector_type(4)))  uint32_t u32x4;
typedef __attribute__((ext_vector_type(8)))  int      i32x8;
typedef __attribute__((ext_vector_type(4)))  int      i32x4;

#define B_   512
#define I_   128
#define T_   1024
#define H_   89
#define G4   356   // 4*H
#define GP   368   // 4*H padded to 16-tile multiple (23 tiles)
#define KP   96    // H padded to 32 multiple (3 WMMA k-steps)
#define O_   64
#define NTILES 23  // GP/16

// ----------------------- TDM descriptor helpers -----------------------------
__device__ __forceinline__ void tdm_load_tile(uint32_t lds_addr, uint64_t gaddr,
                                              const i32x8& g1) {
  // D# group0: count=1, lds_addr, 57-bit global addr, type=2 (ISA 8.3)
  u32x4 g0;
  g0[0] = 1u;
  g0[1] = lds_addr;
  g0[2] = (uint32_t)gaddr;
  g0[3] = ((uint32_t)(gaddr >> 32) & 0x01FFFFFFu) | 0x80000000u;
  i32x4 z4 = {0, 0, 0, 0};                      // groups 2/3 unused (2D tile)
  i32x8 z8 = {0, 0, 0, 0, 0, 0, 0, 0};
  __builtin_amdgcn_tensor_load_to_lds(g0, g1, z4, z4, z8, 0);
}

__device__ __forceinline__ void tdm_store_1d(uint32_t lds_addr, uint64_t gaddr,
                                             int nelem2B) {
  // 1-D tile of nelem2B bf16 elements, LDS -> global (contiguous block).
  u32x4 g0;
  g0[0] = 1u;
  g0[1] = lds_addr;
  g0[2] = (uint32_t)gaddr;
  g0[3] = ((uint32_t)(gaddr >> 32) & 0x01FFFFFFu) | 0x80000000u;
  i32x8 g1;
  const uint32_t td0 = 1u << 20;                // generous OOB bound
  g1[0] = (int)(1u << 16);                      // data_size = 2 bytes
  g1[1] = (int)((td0 & 0xFFFFu) << 16);         // tensor_dim0 lo16 (=0)
  g1[2] = (int)(td0 >> 16);                     // tensor_dim0 hi16; td1 lo16=0
  g1[3] = (int)(((uint32_t)nelem2B) << 16);     // td1 hi16=0 | tile_dim0
  g1[4] = 0;                                    // tile_dim1=0 (unused), dim2=0
  g1[5] = 0;                                    // strides unused for 1-D
  g1[6] = 0;
  g1[7] = 0;
  i32x4 z4 = {0, 0, 0, 0};
  i32x8 z8 = {0, 0, 0, 0, 0, 0, 0, 0};
  __builtin_amdgcn_tensor_store_from_lds(g0, g1, z4, z4, z8, 0);
}

// -------------------------------- prep -------------------------------------
__global__ __launch_bounds__(256) void lstm_prep_kernel(
    const float* __restrict__ W_ih, const float* __restrict__ W_hh,
    const float* __restrict__ b_ih, const float* __restrict__ b_hh,
    bf16_t* __restrict__ wih, bf16_t* __restrict__ whh, float* __restrict__ bias) {
  const int tid  = blockIdx.x * blockDim.x + threadIdx.x;
  const int nthr = gridDim.x * blockDim.x;
  for (int j = tid; j < GP * I_; j += nthr) {
    int g = j >> 7, i = j & (I_ - 1);
    wih[j] = (bf16_t)((g < G4) ? W_ih[g * I_ + i] : 0.f);
  }
  for (int j = tid; j < GP * KP; j += nthr) {
    int g = j / KP, k = j - g * KP;
    whh[j] = (bf16_t)((g < G4 && k < H_) ? W_hh[g * H_ + k] : 0.f);
  }
  for (int j = tid; j < GP; j += nthr)
    bias[j] = (j < G4) ? (b_ih[j] + b_hh[j]) : 0.f;
}

// ---------------------------- input GEMM (xg) -------------------------------
// grid: B * (T/16) blocks, 256 threads (8 waves). Each WG: one b, 16 t's.
__global__ __launch_bounds__(256) void lstm_gates_gemm_kernel(
    const float* __restrict__ x, const bf16_t* __restrict__ wih,
    const float* __restrict__ bias, bf16_t* __restrict__ xg) {
  __shared__ bf16_t sA[16][I_];                 // [t][i] bf16 tile, 4 KB
  __shared__ bf16_t s_out[16][GP];              // staged output tile, 11.5 KB
  const int b   = blockIdx.x >> 6;              // T/16 == 64
  const int t0  = (blockIdx.x & 63) << 4;
  const int tid = threadIdx.x;

  // stage x[b, i, t0..t0+15] -> sA[t][i] (fp32 -> bf16, transpose)
  {
    const int i  = tid >> 1;
    const int tp = (tid & 1) * 8;
    const float* src = x + ((size_t)b * I_ + i) * T_ + t0 + tp;
    float4 a0 = *(const float4*)(src);
    float4 a1 = *(const float4*)(src + 4);
    sA[tp + 0][i] = (bf16_t)a0.x; sA[tp + 1][i] = (bf16_t)a0.y;
    sA[tp + 2][i] = (bf16_t)a0.z; sA[tp + 3][i] = (bf16_t)a0.w;
    sA[tp + 4][i] = (bf16_t)a1.x; sA[tp + 5][i] = (bf16_t)a1.y;
    sA[tp + 6][i] = (bf16_t)a1.z; sA[tp + 7][i] = (bf16_t)a1.w;
  }
  __syncthreads();

  const int lane = tid & 31;
  const int wid  = tid >> 5;
  const int m    = lane & 15;
  const int half = lane >> 4;

  // A fragments, 4 k-steps of 32 over I=128 (contiguous k = 16*half + e per lane)
  v16bf A[4];
  for (int ks = 0; ks < 4; ++ks) {
    const bf16_t* row = &sA[m][0];
    v8bf lo = *(const v8bf*)(row + ks * 32 + half * 16);
    v8bf hi = *(const v8bf*)(row + ks * 32 + half * 16 + 8);
    for (int e = 0; e < 8; ++e) { A[ks][e] = lo[e]; A[ks][8 + e] = hi[e]; }
  }

  for (int s = 0; s < 3; ++s) {
    const int nt = wid + 8 * s;                 // waves 0..7 cover 23 n-tiles
    if (nt >= NTILES) break;
    const int g = nt * 16 + m;                  // lane's N column
    v8f acc = {};
    for (int ks = 0; ks < 4; ++ks) {
      // B fragment: 16 consecutive K per lane (32B) from bf16 W_ih[g][*]
      v16bf Bf = *(const v16bf*)(wih + g * I_ + ks * 32 + half * 16);
      acc = __builtin_amdgcn_wmma_f32_16x16x32_bf16(
          false, A[ks], false, Bf, (short)0, acc, false, false);
    }
    const float bv = bias[g];
    for (int r = 0; r < 8; ++r)                 // D: M = r + 8*half -> LDS
      s_out[r + half * 8][g] = (bf16_t)(acc[r] + bv);
  }
  __syncthreads();

  // whole tile xg[b, t0..t0+15, :] is contiguous (16*368 bf16 = 11776 B):
  // bulk move LDS -> HBM via TDM; s_endpgm's implicit wait-idle covers it.
  if (wid == 0) {
    tdm_store_1d((uint32_t)(uintptr_t)&s_out[0][0],
                 (uint64_t)(uintptr_t)xg + ((uint64_t)b * T_ + t0) * GP * 2u,
                 16 * GP);
  }
}

// ------------------------------- scan kernel --------------------------------
// grid: B/16 = 32 blocks, 256 threads (8 waves). Each WG owns 16 batch rows.
__global__ __launch_bounds__(256) void lstm_scan_kernel(
    const bf16_t* __restrict__ xg, const bf16_t* __restrict__ whh,
    const float* __restrict__ W_out, const float* __restrict__ b_out,
    float* __restrict__ out) {
  __shared__ bf16_t s_h[16][KP];                //  3 KB  h (bf16, WMMA A)
  __shared__ float  s_hf[16][KP];               //  6 KB  h (f32, final proj)
  __shared__ float  s_c[16][KP];                //  6 KB  cell state
  __shared__ float  s_g[16][GP];                // 23 KB  gate pre-activations
  __shared__ bf16_t s_xg[2][16][GP];            // 23 KB  TDM double buffer

  const int b0   = blockIdx.x * 16;
  const int tid  = threadIdx.x;
  const int lane = tid & 31;
  const int wid  = tid >> 5;
  const int m    = lane & 15;
  const int half = lane >> 4;

  for (int j = tid; j < 16 * KP; j += 256) {    // zero h (incl. K pad) and c
    ((bf16_t*)s_h)[j] = (bf16_t)0.f;
    ((float*)s_hf)[j] = 0.f;
    ((float*)s_c)[j]  = 0.f;
  }

  // D# group1 constants (ISA 8.4): data_size=2B, tensor_dim0=GP,
  // tensor_dim1=1<<20 (large), tile = GP x 16 rows, row stride = T*GP elems.
  i32x8 g1;
  {
    const uint32_t td1 = 1u << 20;
    const uint32_t str = (uint32_t)(T_ * GP);
    g1[0] = (int)(1u << 16);                        // data_size = 2 bytes
    g1[1] = (int)(((uint32_t)GP & 0xFFFFu) << 16);  // tensor_dim0 lo16
    g1[2] = (int)((td1 & 0xFFFFu) << 16);           // td0 hi16=0 | td1 lo16
    g1[3] = (int)((td1 >> 16) | (((uint32_t)GP) << 16)); // td1 hi | tile_dim0
    g1[4] = 16;                                     // tile_dim1=16, tile_dim2=0
    g1[5] = (int)str;                               // tensor_dim0_stride lo32
    g1[6] = 0;                                      // stride hi | dim1_stride lo
    g1[7] = 0;
  }
  const uint64_t xg_base =
      (uint64_t)(uintptr_t)xg + ((uint64_t)b0 * T_) * GP * 2u;
  const uint32_t lds0 = (uint32_t)(uintptr_t)&s_xg[0][0][0];
  const uint32_t lds1 = (uint32_t)(uintptr_t)&s_xg[1][0][0];

  if (wid == 0) tdm_load_tile(lds0, xg_base, g1);        // prefetch t=0

  for (int t = 0; t < T_; ++t) {
    if (wid == 0) {
      if (t + 1 < T_) {                         // prefetch t+1 into other buf
        tdm_load_tile(((t + 1) & 1) ? lds1 : lds0,
                      xg_base + (uint64_t)(t + 1) * GP * 2u, g1);
        __builtin_amdgcn_s_wait_tensorcnt(1);   // tile t complete
      } else {
        __builtin_amdgcn_s_wait_tensorcnt(0);
      }
    }
    __syncthreads();  // tile t visible; prev elementwise h/c writes visible

    // ---- gates = xg(t) + h . W_hh^T  (WMMA, K=96, N=368, M=16) ----
    v16bf A[3];
    for (int ks = 0; ks < 3; ++ks) {
      const bf16_t* row = &s_h[m][0];
      v8bf lo = *(const v8bf*)(row + ks * 32 + half * 16);
      v8bf hi = *(const v8bf*)(row + ks * 32 + half * 16 + 8);
      for (int e = 0; e < 8; ++e) { A[ks][e] = lo[e]; A[ks][8 + e] = hi[e]; }
    }
    const bf16_t* xrow = &s_xg[t & 1][0][0];
    for (int s = 0; s < 3; ++s) {
      const int nt = wid + 8 * s;
      if (nt < NTILES) {
        const int g = nt * 16 + m;
        v8f acc;
        for (int r = 0; r < 8; ++r)             // C init = xg tile (bf16->f32)
          acc[r] = (float)xrow[(size_t)(r + half * 8) * GP + g];
        for (int ks = 0; ks < 3; ++ks) {
          v16bf Bf = *(const v16bf*)(whh + g * KP + ks * 32 + half * 16);
          acc = __builtin_amdgcn_wmma_f32_16x16x32_bf16(
              false, A[ks], false, Bf, (short)0, acc, false, false);
        }
        for (int r = 0; r < 8; ++r) s_g[r + half * 8][g] = acc[r];
      }
    }
    __syncthreads();

    // ---- elementwise LSTM cell: i,f,g,o at offsets 0,H,2H,3H ----
    for (int j = tid; j < 16 * KP; j += 256) {
      const int mm = j / KP;
      const int k  = j - mm * KP;
      if (k < H_) {
        float gi = s_g[mm][k];
        float gf = s_g[mm][H_ + k];
        float gg = s_g[mm][2 * H_ + k];
        float go = s_g[mm][3 * H_ + k];
        gi = 1.f / (1.f + __expf(-gi));
        gf = 1.f / (1.f + __expf(-gf));
        go = 1.f / (1.f + __expf(-go));
        const float tg = 1.f - 2.f / (__expf(2.f * gg) + 1.f);   // tanh
        const float c  = gf * s_c[mm][k] + gi * tg;
        s_c[mm][k] = c;
        const float th = 1.f - 2.f / (__expf(2.f * c) + 1.f);    // tanh
        const float h  = go * th;
        s_hf[mm][k] = h;
        s_h[mm][k]  = (bf16_t)h;
      }
    }
  }
  __syncthreads();

  // ---- fused output projection: out = hT . W_out^T + b_out  (16x64) ----
  for (int j = tid; j < 16 * O_; j += 256) {
    const int mm = j >> 6, o = j & 63;
    float acc = b_out[o];
    const float* wr = W_out + o * H_;
    for (int k = 0; k < H_; ++k) acc += s_hf[mm][k] * wr[k];
    out[(size_t)(b0 + mm) * O_ + o] = acc;
  }
}

// ------------------------------- launcher -----------------------------------
extern "C" void kernel_launch(void* const* d_in, const int* in_sizes, int n_in,
                              void* d_out, int out_size, void* d_ws, size_t ws_size,
                              hipStream_t stream) {
  const float* x     = (const float*)d_in[0];   // [512,128,1024]
  const float* W_ih  = (const float*)d_in[1];   // [356,128]
  const float* W_hh  = (const float*)d_in[2];   // [356,89]
  const float* b_ih  = (const float*)d_in[3];   // [356]
  const float* b_hh  = (const float*)d_in[4];   // [356]
  const float* W_out = (const float*)d_in[5];   // [64,89]
  const float* b_out = (const float*)d_in[6];   // [64]
  float* out = (float*)d_out;                   // [512,64]

  char* ws = (char*)d_ws;
  const size_t xg_bytes  = (size_t)B_ * T_ * GP * 2;   // ~386 MB bf16
  const size_t wih_bytes = (size_t)GP * I_ * 2;
  const size_t whh_bytes = (size_t)GP * KP * 2;
  bf16_t* xg   = (bf16_t*)ws;
  bf16_t* wih  = (bf16_t*)(ws + xg_bytes);
  bf16_t* whh  = (bf16_t*)(ws + xg_bytes + wih_bytes);
  float*  bias = (float*)(ws + xg_bytes + wih_bytes + whh_bytes);
  (void)in_sizes; (void)n_in; (void)out_size; (void)ws_size;

  lstm_prep_kernel<<<64, 256, 0, stream>>>(W_ih, W_hh, b_ih, b_hh, wih, whh, bias);
  lstm_gates_gemm_kernel<<<B_ * (T_ / 16), 256, 0, stream>>>(x, wih, bias, xg);
  lstm_scan_kernel<<<B_ / 16, 256, 0, stream>>>(xg, whh, W_out, b_out, out);
}
